// ImprovedSpiralNet_54606214201421
// MI455X (gfx1250) — matrix operand order repaired and checked by reading, not compiled, not run on hardware
//
#include <hip/hip_runtime.h>

typedef __attribute__((ext_vector_type(16))) __bf16 v16bf;
typedef __attribute__((ext_vector_type(8)))  __bf16 v8bf;
typedef __attribute__((ext_vector_type(8)))  float  v8f;
typedef __attribute__((ext_vector_type(4)))  int    v4i;
typedef v4i __attribute__((address_space(1))) as1_v4i;
typedef v4i __attribute__((address_space(3))) as3_v4i;

#define EPSV 1e-5f

#if defined(__HIP_DEVICE_COMPILE__) && \
    __has_builtin(__builtin_amdgcn_global_load_async_to_lds_b128)
#define USE_ASYNC 1
#else
#define USE_ASYNC 0
#endif

template <int N>
__device__ __forceinline__ void wait_async() {
#if defined(__HIP_DEVICE_COMPILE__) && \
    __has_builtin(__builtin_amdgcn_s_wait_asynccnt)
  __builtin_amdgcn_s_wait_asynccnt(N);
#elif USE_ASYNC
  asm volatile("s_wait_asynccnt %0" ::"i"(N) : "memory");
#endif
}

// ---------------------------------------------------------------------------
// Weight cast f32 -> bf16 with K-permute: in[o][c*L + l] -> out[o][l*Cc + c]
// (L==1 gives an identity cast for the skip weights)
// ---------------------------------------------------------------------------
__global__ void k_cast_w(const float* __restrict__ w, __bf16* __restrict__ wp,
                         int O, int Cc, int L) {
  int t = blockIdx.x * blockDim.x + threadIdx.x;
  int n = O * Cc * L;
  if (t >= n) return;
  int o = t / (Cc * L);
  int r = t % (Cc * L);
  int c = r / L;
  int l = r % L;
  wp[(size_t)o * Cc * L + (size_t)l * Cc + c] = (__bf16)w[t];
}

// ---------------------------------------------------------------------------
// Encoder block 0: 3-channel spiral conv (K=36), scalar VALU (trivial FLOPs)
// ---------------------------------------------------------------------------
__global__ __launch_bounds__(256) void k_conv0(
    const float* __restrict__ verts, const int* __restrict__ idx,
    const float* __restrict__ w, const float* __restrict__ bias,
    float* __restrict__ y, int V) {
  int v = blockIdx.x * 256 + threadIdx.x;
  int o = blockIdx.y;
  int b = blockIdx.z;
  __shared__ float sw[36];
  if (threadIdx.x < 36) sw[threadIdx.x] = w[o * 36 + threadIdx.x];
  __syncthreads();
  float acc = bias[o];
#pragma unroll
  for (int l = 0; l < 12; ++l) {
    int j = idx[(size_t)v * 12 + l];
    const float* p = verts + ((size_t)b * V + j) * 3;
    acc += sw[0 * 12 + l] * p[0] + sw[1 * 12 + l] * p[1] + sw[2 * 12 + l] * p[2];
  }
  y[((size_t)b * 256 + o) * V + v] = acc;
}

// ---------------------------------------------------------------------------
// BN batch statistics per channel over (B,V): mean + inv-std
// ---------------------------------------------------------------------------
__global__ __launch_bounds__(256) void k_bn_stats(
    const float* __restrict__ y, float* __restrict__ mean,
    float* __restrict__ istd, int C, int V, int B) {
  int c = blockIdx.x;
  int tid = threadIdx.x;
  float s = 0.f, s2 = 0.f;
  for (int b = 0; b < B; ++b) {
    const float* p = y + ((size_t)b * C + c) * V;
    for (int v = tid; v < V; v += 256) {
      float x = p[v];
      s += x;
      s2 += x * x;
    }
  }
  __shared__ float rs[256], rs2[256];
  rs[tid] = s;
  rs2[tid] = s2;
  __syncthreads();
  for (int st = 128; st > 0; st >>= 1) {
    if (tid < st) { rs[tid] += rs[tid + st]; rs2[tid] += rs2[tid + st]; }
    __syncthreads();
  }
  if (tid == 0) {
    float n = (float)(B * V);
    float m = rs[0] / n;
    float var = rs2[0] / n - m * m;
    mean[c] = m;
    istd[c] = rsqrtf(var + EPSV);
  }
}

// ---------------------------------------------------------------------------
// y = relu((y-mean)*istd*g + beta) [+ residual]; optional per-(b,c) mean for SE
// ---------------------------------------------------------------------------
__global__ __launch_bounds__(256) void k_bn_apply(
    float* __restrict__ y, const float* __restrict__ xres,
    const float* __restrict__ mean, const float* __restrict__ istd,
    const float* __restrict__ g, const float* __restrict__ beta,
    float* __restrict__ sbc, int C, int V) {
  int c = blockIdx.x, b = blockIdx.y, tid = threadIdx.x;
  size_t base = ((size_t)b * C + c) * V;
  float m = mean[c], is = istd[c] * g[c], bt = beta[c];
  float s = 0.f;
  for (int v = tid; v < V; v += 256) {
    float val = (y[base + v] - m) * is + bt;
    val = fmaxf(val, 0.f);
    if (xres) val += xres[base + v];
    y[base + v] = val;
    s += val;
  }
  if (sbc) {
    __shared__ float rs[256];
    rs[tid] = s;
    __syncthreads();
    for (int st = 128; st > 0; st >>= 1) {
      if (tid < st) rs[tid] += rs[tid + st];
      __syncthreads();
    }
    if (tid == 0) sbc[b * C + c] = rs[0] / (float)V;
  }
}

// ---------------------------------------------------------------------------
// Squeeze-Excitation: scale[b][c] = sigmoid(W2 relu(W1 s[b]))  (tiny)
// ---------------------------------------------------------------------------
__global__ __launch_bounds__(256) void k_se(
    const float* __restrict__ sbc, const float* __restrict__ w1,
    const float* __restrict__ w2, float* __restrict__ scale, int B) {
  __shared__ float s[256], h[32];
  int tid = threadIdx.x;
  for (int b = 0; b < B; ++b) {
    s[tid] = sbc[b * 256 + tid];
    __syncthreads();
    if (tid < 32) {
      float a = 0.f;
      for (int c = 0; c < 256; ++c) a += w1[tid * 256 + c] * s[c];
      h[tid] = fmaxf(a, 0.f);
    }
    __syncthreads();
    float a = 0.f;
#pragma unroll
    for (int r = 0; r < 32; ++r) a += w2[tid * 32 + r] * h[r];
    scale[b * 256 + tid] = 1.f / (1.f + __expf(-a));
    __syncthreads();
  }
}

// ---------------------------------------------------------------------------
// Scale (optional) + transpose/cast: src (B,256,V) f32 -> dxt (B,V,256) bf16
// optionally also write scaled f32 copy to dx. 32x32 LDS tile transpose.
// ---------------------------------------------------------------------------
__global__ __launch_bounds__(256) void k_xpose(
    const float* __restrict__ src, const float* __restrict__ scale,
    float* __restrict__ dx, __bf16* __restrict__ dxt, int V) {
  __shared__ float t[32][33];
  int b = blockIdx.z, c0 = blockIdx.y * 32, v0 = blockIdx.x * 32;
  int tx = threadIdx.x & 31, ty = threadIdx.x >> 5;
#pragma unroll
  for (int i = 0; i < 32; i += 8) {
    int c = c0 + ty + i;
    float val = src[((size_t)b * 256 + c) * V + v0 + tx];
    if (scale) val *= scale[b * 256 + c];
    if (dx) dx[((size_t)b * 256 + c) * V + v0 + tx] = val;
    t[ty + i][tx] = val;
  }
  __syncthreads();
#pragma unroll
  for (int i = 0; i < 32; i += 8) {
    dxt[((size_t)b * V + v0 + ty + i) * 256 + c0 + tx] = (__bf16)t[tx][ty + i];
  }
}

// ---------------------------------------------------------------------------
// Spiral-conv GEMM via WMMA bf16:
//   y(256,V) = Wp(256,3072) x G(3072,V) + bias, G[k][v]=xt[b][idx[v][l]][c],
//   k = l*256 + c (weights pre-permuted to match).
// Tile: M=256 x N=64 per workgroup; 8 waves, each 2 M-tiles x 4 N-tiles
// (8 v_wmma per K-step). K split into 24 chunks of 128, double-buffered in
// LDS (2 x 16 KB), filled with GLOBAL_LOAD_ASYNC_TO_LDS_B128 (ASYNCcnt) so
// the gather for chunk ch+1 overlaps the WMMAs of chunk ch.
// ---------------------------------------------------------------------------
__global__ __launch_bounds__(256) void k_spiral_gemm(
    const __bf16* __restrict__ xt, const int* __restrict__ idx,
    const __bf16* __restrict__ wp, const float* __restrict__ bias,
    float* __restrict__ y, int V) {
  constexpr int Cc = 256, L = 12, K = Cc * L;
  constexpr int NT = 64;                   // vertices per workgroup
  constexpr int KC = 128;                  // k per chunk (half a tap)
  constexpr int NCH = K / KC;              // 24 chunks
  constexpr int IPT = NT * KC / 8 / 256;   // b128 issues per thread per chunk = 4
  __shared__ __bf16 gl[2][NT * KC];        // 2 x 16 KB double buffer [v][k]
  __shared__ int sidx[NT * L];
  int b = blockIdx.y;
  int v0 = blockIdx.x * NT;
  int tid = threadIdx.x;
  int wave = tid >> 5, lane = tid & 31;
  int mrow = lane & 15, hi = lane >> 4;
  int m_base = wave * 32;
  v8f acc[2][4] = {};

  for (int t = tid; t < NT * L; t += 256)
    sidx[t] = idx[(size_t)(v0 + t / L) * L + (t % L)];
  __syncthreads();

  auto issue = [&](int ch, int buf) {
    int l = ch >> 1, h = ch & 1;
#pragma unroll
    for (int i = 0; i < IPT; ++i) {
      int e = tid + i * 256;
      int v = e >> 4;       // 16 b128 per vertex row (128 k)
      int c8 = e & 15;
      int j = sidx[v * L + l];
      const __bf16* gsrc = &xt[((size_t)b * V + j) * Cc + h * 128 + c8 * 8];
      __bf16* ldst = &gl[buf][v * KC + c8 * 8];
#if USE_ASYNC
      __builtin_amdgcn_global_load_async_to_lds_b128((as1_v4i*)gsrc,
                                                     (as3_v4i*)ldst, 0, 0);
#else
      *(v8bf*)ldst = *(const v8bf*)gsrc;
#endif
    }
  };

  issue(0, 0);
  for (int ch = 0; ch < NCH; ++ch) {
    int cur = ch & 1;
    if (ch + 1 < NCH) {
      issue(ch + 1, cur ^ 1);  // prefetch next chunk (prev reads done: bottom barrier)
#if USE_ASYNC
      wait_async<IPT>();       // in-order returns: current chunk has landed
#endif
    } else {
#if USE_ASYNC
      wait_async<0>();
#endif
    }
    __syncthreads();
    int kgbase = (ch >> 1) * 256 + (ch & 1) * 128;
    for (int ks = 0; ks < KC; ks += 32) {
      int kg = kgbase + ks;
      v16bf a[2], bm[4];
#pragma unroll
      for (int mt = 0; mt < 2; ++mt) {
        // 16-bit A fragment: lanes<16 hold K 0..15; lanes>=16 hold 16..31
        const __bf16* wr = wp + (size_t)(m_base + mt * 16 + mrow) * K + kg;
        v8bf lo = *(const v8bf*)(wr + hi * 16);
        v8bf hv = *(const v8bf*)(wr + hi * 16 + 8);
#pragma unroll
        for (int i = 0; i < 8; ++i) { a[mt][i] = lo[i]; a[mt][8 + i] = hv[i]; }
      }
#pragma unroll
      for (int nt = 0; nt < 4; ++nt)  // B: lanes<16 K 0..15, lanes>=16 K 16..31
        bm[nt] = *(const v16bf*)&gl[cur][(nt * 16 + mrow) * KC + ks + hi * 16];
#pragma unroll
      for (int mt = 0; mt < 2; ++mt)
#pragma unroll
        for (int nt = 0; nt < 4; ++nt)
          acc[mt][nt] = __builtin_amdgcn_wmma_f32_16x16x32_bf16(
              false, a[mt], false, bm[nt], (short)0, acc[mt][nt], false, false);
    }
    __syncthreads();  // release current buffer for refill
  }
  // Epilogue: C/D layout — VGPR r: lanes<16 row r, lanes>=16 row r+8
#pragma unroll
  for (int mt = 0; mt < 2; ++mt)
#pragma unroll
    for (int nt = 0; nt < 4; ++nt) {
      int col = v0 + nt * 16 + mrow;
#pragma unroll
      for (int r = 0; r < 8; ++r) {
        int row = m_base + mt * 16 + r + hi * 8;
        y[((size_t)b * 256 + row) * V + col] = acc[mt][nt][r] + bias[row];
      }
    }
}

// ---------------------------------------------------------------------------
// Skip GEMM via WMMA bf16: y(256,V) = Wp(256,512) x [x; feat](512,V) + bias
// Both halves come from bf16-transposed activations (B,V,256).
// ---------------------------------------------------------------------------
__global__ __launch_bounds__(256) void k_skip_gemm(
    const __bf16* __restrict__ xta, const __bf16* __restrict__ xtb,
    const __bf16* __restrict__ wp, const float* __restrict__ bias,
    float* __restrict__ y, int V) {
  constexpr int Cc = 256, K = 512;
  __shared__ __bf16 gl[32 * K];  // 32 KB
  int b = blockIdx.y, v0 = blockIdx.x * 32, tid = threadIdx.x;
  int wave = tid >> 5, lane = tid & 31;
  int mrow = lane & 15, hi = lane >> 4;
  int m_base = wave * 32;
  v8f acc[2][2] = {};
  for (int e8 = tid; e8 < 32 * K / 8; e8 += 256) {
    int v = e8 / (K / 8);
    int r = e8 % (K / 8);
    int half = r / (Cc / 8);
    int c8 = r % (Cc / 8);
    const __bf16* src = (half ? xtb : xta) + ((size_t)b * V + v0 + v) * Cc + c8 * 8;
    *(v8bf*)&gl[v * K + half * Cc + c8 * 8] = *(const v8bf*)src;
  }
  __syncthreads();
  for (int ks = 0; ks < K; ks += 32) {
    v16bf a[2], bm[2];
#pragma unroll
    for (int mt = 0; mt < 2; ++mt) {
      const __bf16* wr = wp + (size_t)(m_base + mt * 16 + mrow) * K + ks;
      v8bf lo = *(const v8bf*)(wr + hi * 16);
      v8bf hv = *(const v8bf*)(wr + hi * 16 + 8);
#pragma unroll
      for (int i = 0; i < 8; ++i) { a[mt][i] = lo[i]; a[mt][8 + i] = hv[i]; }
    }
#pragma unroll
    for (int nt = 0; nt < 2; ++nt)
      bm[nt] = *(const v16bf*)&gl[(nt * 16 + mrow) * K + ks + hi * 16];
#pragma unroll
    for (int mt = 0; mt < 2; ++mt)
#pragma unroll
      for (int nt = 0; nt < 2; ++nt)
        acc[mt][nt] = __builtin_amdgcn_wmma_f32_16x16x32_bf16(
            false, a[mt], false, bm[nt], (short)0, acc[mt][nt], false, false);
  }
#pragma unroll
  for (int mt = 0; mt < 2; ++mt)
#pragma unroll
    for (int nt = 0; nt < 2; ++nt) {
      int col = v0 + nt * 16 + mrow;
#pragma unroll
      for (int r = 0; r < 8; ++r) {
        int row = m_base + mt * 16 + r + hi * 8;
        y[((size_t)b * 256 + row) * V + col] = acc[mt][nt][r] + bias[row];
      }
    }
}

// ---------------------------------------------------------------------------
// Dense 1x1 conv (classifier): out[b][o][v] = bias[o] + sum_c w[o][c] x[b][c][v]
// ---------------------------------------------------------------------------
__global__ __launch_bounds__(256) void k_lin(
    const float* __restrict__ x, const float* __restrict__ w,
    const float* __restrict__ bias, float* __restrict__ out, int Cin, int V) {
  int v = blockIdx.x * 256 + threadIdx.x;
  int o = blockIdx.y, b = blockIdx.z;
  __shared__ float sw[256];
  for (int c = threadIdx.x; c < Cin; c += 256) sw[c] = w[o * Cin + c];
  __syncthreads();
  float acc = bias[o];
  for (int c = 0; c < Cin; ++c)
    acc += sw[c] * x[((size_t)b * Cin + c) * V + v];
  out[((size_t)b * gridDim.y + o) * V + v] = acc;
}

// ---------------------------------------------------------------------------
extern "C" void kernel_launch(void* const* d_in, const int* in_sizes, int n_in,
                              void* d_out, int out_size, void* d_ws,
                              size_t ws_size, hipStream_t stream) {
  (void)in_sizes; (void)n_in; (void)out_size; (void)ws_size;
  const int B = 2, V = 16384, C = 256, L = 12, K = C * L;

  const float* verts    = (const float*)d_in[0];
  const int*   spir     = (const int*)d_in[1];
  const float* enc0_w   = (const float*)d_in[2];
  const float* enc0_b   = (const float*)d_in[3];
  const float* enc0_g   = (const float*)d_in[4];
  const float* enc0_be  = (const float*)d_in[5];
  const float* enc_w    = (const float*)d_in[6];
  const float* enc_b    = (const float*)d_in[7];
  const float* enc_g    = (const float*)d_in[8];
  const float* enc_be   = (const float*)d_in[9];
  const float* se_w1    = (const float*)d_in[10];
  const float* se_w2    = (const float*)d_in[11];
  const float* skip_w   = (const float*)d_in[12];
  const float* skip_b   = (const float*)d_in[13];
  const float* dec_w    = (const float*)d_in[14];
  const float* dec_b    = (const float*)d_in[15];
  const float* dec_g    = (const float*)d_in[16];
  const float* dec_be   = (const float*)d_in[17];
  const float* cls1_w   = (const float*)d_in[18];
  const float* cls1_b   = (const float*)d_in[19];
  const float* cls1_g   = (const float*)d_in[20];
  const float* cls1_be  = (const float*)d_in[21];
  const float* cls2_w   = (const float*)d_in[22];
  const float* cls2_b   = (const float*)d_in[23];
  const float* cls2_g   = (const float*)d_in[24];
  const float* cls2_be  = (const float*)d_in[25];
  const float* cls3_w   = (const float*)d_in[26];
  const float* cls3_b   = (const float*)d_in[27];

  char* ws = (char*)d_ws;
  size_t off = 0;
  auto alloc = [&](size_t bytes) -> void* {
    void* p = ws + off;
    off = (off + bytes + 255) & ~(size_t)255;
    return p;
  };

  __bf16* wp_enc  = (__bf16*)alloc(3 * (size_t)C * K * 2);
  __bf16* wp_dec  = (__bf16*)alloc(3 * (size_t)C * K * 2);
  __bf16* wp_skip = (__bf16*)alloc(3 * (size_t)C * 512 * 2);
  float*  buf_y   = (float*)alloc((size_t)B * C * V * 4);
  float*  buf_x   = (float*)alloc((size_t)B * C * V * 4);
  __bf16* buf_xt  = (__bf16*)alloc((size_t)B * V * C * 2);
  __bf16* feat_t[3];
  for (int i = 0; i < 3; ++i) feat_t[i] = (__bf16*)alloc((size_t)B * V * C * 2);
  float* buf_h1 = (float*)alloc((size_t)B * 128 * V * 4);
  float* buf_h2 = (float*)alloc((size_t)B * 64 * V * 4);
  float* mean   = (float*)alloc(256 * 4);
  float* istd   = (float*)alloc(256 * 4);
  float* sbc    = (float*)alloc((size_t)B * 256 * 4);
  float* scl    = (float*)alloc((size_t)B * 256 * 4);

  // ---- weight prep (cheap; per call for graph determinism) ----
  {
    int n = C * K, blk = (n + 255) / 256;
    for (int i = 0; i < 3; ++i) {
      k_cast_w<<<blk, 256, 0, stream>>>(enc_w + (size_t)i * C * K,
                                        wp_enc + (size_t)i * C * K, C, C, L);
      k_cast_w<<<blk, 256, 0, stream>>>(dec_w + (size_t)i * C * K,
                                        wp_dec + (size_t)i * C * K, C, C, L);
    }
    int n2 = C * 512, blk2 = (n2 + 255) / 256;
    for (int i = 0; i < 3; ++i)
      k_cast_w<<<blk2, 256, 0, stream>>>(skip_w + (size_t)i * C * 512,
                                         wp_skip + (size_t)i * C * 512, C, 512, 1);
  }

  // ---- encoder block 0 ----
  k_conv0<<<dim3(V / 256, C, B), 256, 0, stream>>>(verts, spir, enc0_w, enc0_b,
                                                   buf_y, V);
  k_bn_stats<<<C, 256, 0, stream>>>(buf_y, mean, istd, C, V, B);
  k_bn_apply<<<dim3(C, B), 256, 0, stream>>>(buf_y, nullptr, mean, istd,
                                             enc0_g, enc0_be, sbc, C, V);
  k_se<<<1, 256, 0, stream>>>(sbc, se_w1, se_w2, scl, B);
  k_xpose<<<dim3(V / 32, C / 32, B), 256, 0, stream>>>(buf_y, scl, buf_x,
                                                       feat_t[0], V);

  // ---- encoder blocks 1..3 (WMMA spiral conv + residual + SE) ----
  for (int i = 1; i <= 3; ++i) {
    const __bf16* src_xt = feat_t[i - 1];
    __bf16* dst_xt = (i < 3) ? feat_t[i] : buf_xt;
    k_spiral_gemm<<<dim3(V / 64, B), 256, 0, stream>>>(
        src_xt, spir + (size_t)i * V * L, wp_enc + (size_t)(i - 1) * C * K,
        enc_b + (size_t)(i - 1) * C, buf_y, V);
    k_bn_stats<<<C, 256, 0, stream>>>(buf_y, mean, istd, C, V, B);
    k_bn_apply<<<dim3(C, B), 256, 0, stream>>>(
        buf_y, buf_x, mean, istd, enc_g + (size_t)(i - 1) * C,
        enc_be + (size_t)(i - 1) * C, sbc, C, V);
    k_se<<<1, 256, 0, stream>>>(sbc, se_w1 + (size_t)i * 32 * C,
                                se_w2 + (size_t)i * C * 32, scl, B);
    k_xpose<<<dim3(V / 32, C / 32, B), 256, 0, stream>>>(buf_y, scl, buf_x,
                                                         dst_xt, V);
  }

  // ---- decoder (skip GEMM + WMMA spiral conv + residual) ----
  for (int i = 0; i < 3; ++i) {
    const int* idx = spir + (size_t)(2 - i) * V * L;
    k_skip_gemm<<<dim3(V / 32, B), 256, 0, stream>>>(
        buf_xt, feat_t[2 - i], wp_skip + (size_t)i * C * 512,
        skip_b + (size_t)i * C, buf_x, V);
    k_xpose<<<dim3(V / 32, C / 32, B), 256, 0, stream>>>(buf_x, nullptr,
                                                         nullptr, buf_xt, V);
    k_spiral_gemm<<<dim3(V / 64, B), 256, 0, stream>>>(
        buf_xt, idx, wp_dec + (size_t)i * C * K, dec_b + (size_t)i * C, buf_y, V);
    k_bn_stats<<<C, 256, 0, stream>>>(buf_y, mean, istd, C, V, B);
    k_bn_apply<<<dim3(C, B), 256, 0, stream>>>(
        buf_y, buf_x, mean, istd, dec_g + (size_t)i * C,
        dec_be + (size_t)i * C, nullptr, C, V);
    k_xpose<<<dim3(V / 32, C / 32, B), 256, 0, stream>>>(buf_y, nullptr, buf_x,
                                                         buf_xt, V);
  }

  // ---- classifier ----
  k_lin<<<dim3(V / 256, 128, B), 256, 0, stream>>>(buf_x, cls1_w, cls1_b,
                                                   buf_h1, C, V);
  k_bn_stats<<<128, 256, 0, stream>>>(buf_h1, mean, istd, 128, V, B);
  k_bn_apply<<<dim3(128, B), 256, 0, stream>>>(buf_h1, nullptr, mean, istd,
                                               cls1_g, cls1_be, nullptr, 128, V);
  k_lin<<<dim3(V / 256, 64, B), 256, 0, stream>>>(buf_h1, cls2_w, cls2_b,
                                                  buf_h2, 128, V);
  k_bn_stats<<<64, 256, 0, stream>>>(buf_h2, mean, istd, 64, V, B);
  k_bn_apply<<<dim3(64, B), 256, 0, stream>>>(buf_h2, nullptr, mean, istd,
                                              cls2_g, cls2_be, nullptr, 64, V);
  k_lin<<<dim3(V / 256, 1, B), 256, 0, stream>>>(buf_h2, cls3_w, cls3_b,
                                                 (float*)d_out, 64, V);
}
